// old_lidar_DynamicEmbedder_90950227460837
// MI455X (gfx1250) — matrix operand order repaired and checked by reading, not compiled, not run on hardware
//
#include <hip/hip_runtime.h>

typedef __attribute__((ext_vector_type(16))) _Float16 v16h;
typedef __attribute__((ext_vector_type(8)))  _Float16 h8;
typedef __attribute__((ext_vector_type(8)))  float    v8f;
typedef __attribute__((ext_vector_type(4))) unsigned int u32x4;
typedef __attribute__((ext_vector_type(8))) int i32x8;
typedef __attribute__((ext_vector_type(4))) int i32x4;

#define NPTS 120000
#define NSEG (512*512)
#define NB   2

// workspace byte offsets (total ~92 MB)
#define OFF_VID  0
#define OFF_CNT  (960000)                       // NB*NPTS*4
#define OFF_SUM  (OFF_CNT + NB*NSEG*4)
#define OFF_POOL (OFF_SUM + NB*NSEG*3*4)
#define OFF_H0   (OFF_POOL + (size_t)NB*NSEG*32*4)

__device__ __forceinline__ float relu_(float x) { return x > 0.f ? x : 0.f; }

// ---------------- init: zero counts/sums/pooled and the canvas half of d_out
__global__ void k_init(float* zbase, long nz, float* canvas, long nc) {
  long i = (long)blockIdx.x * blockDim.x + threadIdx.x;
  long stride = (long)gridDim.x * blockDim.x;
  for (long j = i; j < nz; j += stride) zbase[j] = 0.f;
  for (long j = i; j < nc; j += stride) canvas[j] = 0.f;
}

// ---------------- pass 1: pillar id, counts, xyz sums
__global__ void k_voxelize(const float* __restrict__ pts, int* __restrict__ vids,
                           int* __restrict__ cnt, float* __restrict__ sums) {
  unsigned i = blockIdx.x * blockDim.x + threadIdx.x;
  if (i >= NB * NPTS) return;
  const float* p = pts + i * 3u;
  float x = p[0], y = p[1], z = p[2];
  int xi = (int)floorf((x - (-51.2f)) / 0.2f);
  int yi = (int)floorf((y - (-51.2f)) / 0.2f);
  xi = min(max(xi, 0), 511);
  yi = min(max(yi, 0), 511);
  int vid = yi * 512 + xi;
  vids[i] = vid;
  unsigned b = i / NPTS;
  unsigned seg = b * (unsigned)NSEG + (unsigned)vid;
  atomicAdd(&cnt[seg], 1);
  atomicAdd(&sums[seg * 3u + 0u], x);
  atomicAdd(&sums[seg * 3u + 1u], y);
  atomicAdd(&sums[seg * 3u + 2u], z);
}

// ---------------- pass 2: 8-dim feats -> Linear(8,32)+BN+ReLU, scatter-max pooled0
__global__ void k_pfn0(const float* __restrict__ pts, const int* __restrict__ vids,
                       const int* __restrict__ cnt, const float* __restrict__ sums,
                       const float* __restrict__ w0, const float* __restrict__ g0,
                       const float* __restrict__ b0, const float* __restrict__ m0,
                       const float* __restrict__ v0,
                       _Float16* __restrict__ h0, int* __restrict__ pool) {
  unsigned i = blockIdx.x * blockDim.x + threadIdx.x;
  if (i >= NB * NPTS) return;
  unsigned b = i / NPTS;
  const float* p = pts + i * 3u;
  float x = p[0], y = p[1], z = p[2];
  int vid = vids[i];
  unsigned seg = b * (unsigned)NSEG + (unsigned)vid;
  float c = fmaxf((float)cnt[seg], 1.f);
  float mx = sums[seg * 3u + 0u] / c;
  float my = sums[seg * 3u + 1u] / c;
  float mz = sums[seg * 3u + 2u] / c;
  int xi = vid & 511, yi = vid >> 9;
  float cx = (xi + 0.5f) * 0.2f - 51.2f;
  float cy = (yi + 0.5f) * 0.2f - 51.2f;
  float f[8] = {x, y, z, x - mx, y - my, z - mz, x - cx, y - cy};
  #pragma unroll 4
  for (int ch = 0; ch < 32; ++ch) {
    float acc = 0.f;
    #pragma unroll
    for (int k = 0; k < 8; ++k) acc = fmaf(f[k], w0[k * 32 + ch], acc);
    float scale = g0[ch] * rsqrtf(v0[ch] + 1e-3f);
    float val = relu_((acc - m0[ch]) * scale + b0[ch]);
    h0[i * 32u + (unsigned)ch] = (_Float16)val;
    // val >= 0: int-max == float-max; pool was zero-initialized
    atomicMax(&pool[seg * 32u + (unsigned)ch], __float_as_int(val));
  }
}

// ---------------- pass 3: concat(h0, pooled0) [N,64] @ w1 [64,64] via WMMA.
// h0 tile arrives via the Tensor Data Mover (TDM) with row padding so LDS rows
// sit at an 80B stride (16B-aligned chunks, conflict-reduced). Fragments are
// built with ds_load_b128 per the ISA 7.12.2 per-lane contiguous K-runs.
__global__ __launch_bounds__(128) void k_pfn1(
    const _Float16* __restrict__ h0, const float* __restrict__ pool,
    const int* __restrict__ vids, const float* __restrict__ w1,
    const float* __restrict__ g1, const float* __restrict__ b1,
    const float* __restrict__ m1, const float* __restrict__ v1,
    float* __restrict__ out) {
  __shared__ _Float16 sh0[64][40];   // 64 pts x 32 feats (+8 pad) <- TDM
  __shared__ _Float16 shp[64][40];   // 64 pts x 32 pooled (+8 pad)
  __shared__ _Float16 sw_t[64][72];  // w1 transposed: [n][k]
  unsigned blk = blockIdx.x;         // NB*1875 blocks
  unsigned b   = blk / 1875u;
  unsigned pt0 = (blk % 1875u) * 64u;
  unsigned tid = threadIdx.x;

  // --- TDM: async load 64x32 f16 tile of h0 into sh0 (wave 0 only) ---
  if (tid < 32) {
    unsigned long long gaddr =
        (unsigned long long)(const void*)(h0 + (size_t)(b * NPTS + pt0) * 32u);
    unsigned int lds_off = (unsigned int)(unsigned long long)(void*)&sh0[0][0];
    u32x4 g0d;
    g0d[0] = 1u;                                   // count = 1 valid descriptor
    g0d[1] = lds_off;                              // lds_addr
    g0d[2] = (unsigned int)(gaddr & 0xFFFFFFFFu);  // global_addr lo
    g0d[3] = (unsigned int)((gaddr >> 32) & 0x1FFFFFFu) | (2u << 30);  // hi + type=2
    i32x8 g1d;
    // data_size=1 (2B), pad_enable, pad_interval=3 (16 DW = one 32-half row),
    // pad_amount=3 (4 DW = 8 halfs) -> LDS row stride 40 halfs
    g1d[0] = (int)((1u << 16) | (1u << 20) | (3u << 22) | (3u << 25));
    g1d[1] = (int)(32u << 16);   // tensor_dim0 = 32 (lo16)
    g1d[2] = (int)(64u << 16);   // tensor_dim0 hi=0; tensor_dim1 = 64 (lo16)
    g1d[3] = (int)(32u << 16);   // tensor_dim1 hi=0; tile_dim0 = 32
    g1d[4] = 64;                 // tile_dim1 = 64, tile_dim2 = 0
    g1d[5] = 32;                 // tensor_dim0_stride = 32
    g1d[6] = 0;
    g1d[7] = 0;
    i32x4 gz = {0, 0, 0, 0};
#if defined(__clang_major__) && (__clang_major__ >= 23)
    i32x8 gz8 = {0, 0, 0, 0, 0, 0, 0, 0};
    __builtin_amdgcn_tensor_load_to_lds(g0d, g1d, gz, gz, gz8, 0);
#else
    __builtin_amdgcn_tensor_load_to_lds(g0d, g1d, gz, gz, 0);
#endif
  }

  // --- w1 -> LDS transposed (f16), vector LDS stores: thread owns column n ---
  {
    unsigned n  = tid & 63u;
    unsigned ks = (tid >> 6) * 32u;  // 0 or 32
    #pragma unroll
    for (unsigned c = 0; c < 4; ++c) {
      h8 v{};
      #pragma unroll
      for (unsigned k = 0; k < 8; ++k)
        v[k] = (_Float16)w1[(ks + c * 8u + k) * 64u + n];
      *(h8*)&sw_t[n][ks + c * 8u] = v;
    }
  }
  // --- pooled0 gather -> shp ---
  for (unsigned e = tid; e < 64u * 32u; e += 128u) {
    unsigned lp = e >> 5, j = e & 31u;
    unsigned gp  = b * NPTS + pt0 + lp;
    unsigned seg = b * (unsigned)NSEG + (unsigned)vids[gp];
    shp[lp][j] = (_Float16)pool[seg * 32u + j];
  }

  if (tid < 32) __builtin_amdgcn_s_wait_tensorcnt((short)0);
  __syncthreads();

  unsigned wave = tid >> 5;
  unsigned lane = tid & 31u;
  unsigned m16  = lane & 15u;
  unsigned hi   = lane >> 4;
  unsigned prow = wave * 16u;

  // A fragments: per-lane contiguous 16B K-runs (ISA 7.12.2 16-bit A 16x32)
  const h8* rlo = (const h8*)&sh0[prow + m16][0];
  const h8* rhi = (const h8*)&shp[prow + m16][0];
  v16h a0 = __builtin_shufflevector(rlo[hi], rlo[2 + hi],
                                    0,1,2,3,4,5,6,7,8,9,10,11,12,13,14,15);
  v16h a1 = __builtin_shufflevector(rhi[hi], rhi[2 + hi],
                                    0,1,2,3,4,5,6,7,8,9,10,11,12,13,14,15);

  // per-lane row bookkeeping for the C/D layout (VGPR r -> M = r + 8*hi)
  unsigned pofs[8];   // point-feature element offsets (fit in 31 bits)
  unsigned vidr[8];
  #pragma unroll
  for (unsigned r = 0; r < 8; ++r) {
    unsigned mm = r + 8u * hi;
    unsigned gp = b * NPTS + pt0 + prow + mm;
    pofs[r] = gp * 64u;
    vidr[r] = (unsigned)vids[gp];
  }

  float* pf = out + (size_t)NB * 64u * NSEG;   // point_feats base (uniform)
  int* canvas_i = (int*)out;
  unsigned cbase = b * 64u;

  #pragma unroll
  for (unsigned t = 0; t < 4; ++t) {
    unsigned n = t * 16u + m16;
    const h8* coln = (const h8*)&sw_t[n][0];   // B: lane n holds K [8*hi..+8) and [16+8*hi..+8)
    v16h bb0 = __builtin_shufflevector(coln[hi], coln[2 + hi],
                                       0,1,2,3,4,5,6,7,8,9,10,11,12,13,14,15);
    v16h bb1 = __builtin_shufflevector(coln[4 + hi], coln[6 + hi],
                                       0,1,2,3,4,5,6,7,8,9,10,11,12,13,14,15);
    v8f acc = {};
    acc = __builtin_amdgcn_wmma_f32_16x16x32_f16(false, a0, false, bb0,
                                                 (short)0, acc, false, false);
    acc = __builtin_amdgcn_wmma_f32_16x16x32_f16(false, a1, false, bb1,
                                                 (short)0, acc, false, false);
    float scale = g1[n] * rsqrtf(v1[n] + 1e-3f);
    float shift = b1[n] - m1[n] * scale;
    unsigned cofs = (cbase + n) * (unsigned)NSEG;   // < 2^31/4 elements
    #pragma unroll
    for (unsigned r = 0; r < 8; ++r) {
      float val = relu_(acc[r] * scale + shift);
      pf[pofs[r] + n] = val;                               // point_feats
      atomicMax(&canvas_i[cofs + vidr[r]], __float_as_int(val));  // canvas max
    }
  }
}

extern "C" void kernel_launch(void* const* d_in, const int* in_sizes, int n_in,
                              void* d_out, int out_size, void* d_ws, size_t ws_size,
                              hipStream_t stream) {
  const float* points = (const float*)d_in[0];
  const float* w0 = (const float*)d_in[1];
  const float* g0 = (const float*)d_in[2];
  const float* b0 = (const float*)d_in[3];
  const float* m0 = (const float*)d_in[4];
  const float* v0 = (const float*)d_in[5];
  const float* w1 = (const float*)d_in[6];
  const float* g1 = (const float*)d_in[7];
  const float* b1 = (const float*)d_in[8];
  const float* m1 = (const float*)d_in[9];
  const float* v1 = (const float*)d_in[10];

  char* ws = (char*)d_ws;
  int*      vids = (int*)(ws + OFF_VID);
  int*      cnt  = (int*)(ws + OFF_CNT);
  float*    sums = (float*)(ws + OFF_SUM);
  float*    pool = (float*)(ws + OFF_POOL);
  _Float16* h0   = (_Float16*)(ws + OFF_H0);
  float*    out  = (float*)d_out;

  long nzero   = (long)(OFF_H0 - OFF_CNT) / 4;   // cnt + sums + pooled0
  long ncanvas = (long)NB * 64 * NSEG;
  k_init<<<2048, 256, 0, stream>>>((float*)(ws + OFF_CNT), nzero, out, ncanvas);

  int npts = NB * NPTS;
  k_voxelize<<<(npts + 255) / 256, 256, 0, stream>>>(points, vids, cnt, sums);
  k_pfn0<<<(npts + 255) / 256, 256, 0, stream>>>(points, vids, cnt, sums,
                                                 w0, g0, b0, m0, v0, h0, (int*)pool);
  k_pfn1<<<NB * 1875, 128, 0, stream>>>(h0, pool, vids, w1, g1, b1, m1, v1, out);
}
